// LSTMEncoder_29652454212153
// MI455X (gfx1250) — matrix operand order repaired and checked by reading, not compiled, not run on hardware
//
#include <hip/hip_runtime.h>
#include <hip/hip_bf16.h>

// LSTM: BATCH=16384, STEP=5, IN_DIM=11, HID=512. Output = final cell state c [16384,512] f32.
// Strategy: 256 blocks x 256 threads (8 waves). Block owns 64 batch rows; h (f16, double
// buffered) and c (f32) live in LDS across all 5 steps. Recurrent GEMM h@W_hh^T done with
// v_wmma_f32_16x16x32_f16; input projection fused as one extra K=32 WMMA slice with
// zero-padded W_ih. W_hh is converted to f16 once (L2-resident, streamed per step).

#define BATCH  16384
#define STEPS  5
#define INDIM  11
#define HID    512
#define G4     2048      // 4*HID
#define MT     64        // rows per block

typedef _Float16 v16h __attribute__((ext_vector_type(16)));
typedef _Float16 v8h  __attribute__((ext_vector_type(8)));
typedef float    v8f  __attribute__((ext_vector_type(8)));

// ---------------- prep: f32->f16 W_hh, padded f16 W_ih[2048][32], fused bias ----------------
__global__ void lstm_prep_kernel(const float* __restrict__ W_ih,
                                 const float* __restrict__ W_hh,
                                 const float* __restrict__ b_ih,
                                 const float* __restrict__ b_hh,
                                 _Float16* __restrict__ whh_h,
                                 _Float16* __restrict__ wih_h,
                                 float* __restrict__ bias) {
  int idx = blockIdx.x * blockDim.x + threadIdx.x;
  if (idx < G4 * HID) whh_h[idx] = (_Float16)W_hh[idx];
  if (idx < G4 * 32) {
    int n = idx >> 5, k = idx & 31;
    wih_h[idx] = (k < INDIM) ? (_Float16)W_ih[n * INDIM + k] : (_Float16)0.0f;
  }
  if (idx < G4) bias[idx] = b_ih[idx] + b_hh[idx];
}

__device__ __forceinline__ float sigmoid_f(float x) {
  return 1.0f / (1.0f + __expf(-x));
}
__device__ __forceinline__ float tanh_f(float x) {
  float e = __expf(2.0f * x);
  return (e - 1.0f) / (e + 1.0f);
}

// A-matrix 16x32 f16 per-lane fragment as used here (contiguous k, same order as B): lane L holds row M=L&15;
// v16h elems 0..15 = K kb+2s..kb+2s+15, s = (L>=16)?8:0.
__device__ __forceinline__ v16h load_a_frag(const _Float16* base, int row_stride,
                                            int row, int kb, int s) {
  const _Float16* p = base + row * row_stride + kb + 2 * s;
  v8h lo = *(const v8h*)(p);
  v8h hi = *(const v8h*)(p + 8);
  return __builtin_shufflevector(lo, hi, 0,1,2,3,4,5,6,7,8,9,10,11,12,13,14,15);
}

// ---------------- main LSTM kernel ----------------
__global__ __launch_bounds__(256, 1)
void lstm_main_kernel(const float* __restrict__ ts,        // [16384][5][11]
                      const _Float16* __restrict__ whh,    // [2048][512] f16
                      const _Float16* __restrict__ wih,    // [2048][32]  f16 (padded)
                      const float* __restrict__ bias,      // [2048] b_ih+b_hh
                      float* __restrict__ out) {           // [16384][512] f32 (c)
  extern __shared__ char smem[];
  _Float16* hs0 = (_Float16*)smem;                    // 64*512 f16 = 64 KB
  _Float16* hs1 = hs0 + MT * HID;                     // 64 KB
  float*    cs  = (float*)(smem + 2 * MT * HID * 2);  // 64*512 f32 = 128 KB
  _Float16* xs  = (_Float16*)(smem + 2 * MT * HID * 2 + MT * HID * 4); // 5*64*32 f16 = 20 KB

  const int tid  = threadIdx.x;
  const int lane = tid & 31;
  const int wave = tid >> 5;
  const int m0   = blockIdx.x * MT;

  // ---- init: padded x (f16) for all 5 steps, and c = 0 ----
  for (int idx = tid; idx < STEPS * MT * 32; idx += 256) {
    int t = idx >> 11;           // /(64*32)
    int r = (idx >> 5) & (MT - 1);
    int k = idx & 31;
    float v = (k < INDIM) ? ts[(size_t)(m0 + r) * (STEPS * INDIM) + t * INDIM + k] : 0.0f;
    xs[idx] = (_Float16)v;
  }
  for (int idx = tid; idx < MT * HID; idx += 256) cs[idx] = 0.0f;
  __syncthreads();

  const int lr  = lane & 15;          // A/B row (M / W_hh-row within tile); C/D column
  const int s   = (lane >> 4) << 3;   // A-fragment K sub-offset (0 or 8)
  const int bk  = (lane >> 4) << 4;   // B-fragment K offset (0 or 16)
  const int mof = (lane >> 4) << 3;   // C/D M offset (0 or 8)
  const v8f vzero = {0.f, 0.f, 0.f, 0.f, 0.f, 0.f, 0.f, 0.f};

  for (int t = 0; t < STEPS; ++t) {
    const _Float16* hcur  = (t & 1) ? hs1 : hs0;
    _Float16*       hnext = (t & 1) ? hs0 : hs1;

    for (int j = 0; j < 4; ++j) {
      const int cg   = wave * 4 + j;   // column group 0..31 (16 cols of H each)
      const int ncol = cg * 16;

      v8f acc[4][4];                   // [gate][row-tile]
      #pragma unroll
      for (int g = 0; g < 4; ++g)
        #pragma unroll
        for (int r = 0; r < 4; ++r) acc[g][r] = vzero;

      // ---- input-projection slice (K=32, W_ih zero-padded) ----
      {
        v16h ax[4];
        #pragma unroll
        for (int r = 0; r < 4; ++r)
          ax[r] = load_a_frag(xs + t * (MT * 32), 32, r * 16 + lr, 0, s);
        #pragma unroll
        for (int g = 0; g < 4; ++g) {
          v16h bx = *(const v16h*)(wih + (size_t)(g * HID + ncol + lr) * 32 + bk);
          #pragma unroll
          for (int r = 0; r < 4; ++r)
            acc[g][r] = __builtin_amdgcn_wmma_f32_16x16x32_f16(
                false, ax[r], false, bx, (short)0, acc[g][r], false, false);
        }
      }

      // ---- recurrent slices: h @ W_hh^T, K=512 (skip at t=0, h=0) ----
      if (t > 0) {
        for (int ks = 0; ks < 16; ++ks) {
          const int kb = ks * 32;
          v16h ah[4];
          #pragma unroll
          for (int r = 0; r < 4; ++r)
            ah[r] = load_a_frag(hcur, HID, r * 16 + lr, kb, s);
          #pragma unroll
          for (int g = 0; g < 4; ++g) {
            v16h bh = *(const v16h*)(whh + (size_t)(g * HID + ncol + lr) * HID + kb + bk);
            #pragma unroll
            for (int r = 0; r < 4; ++r)
              acc[g][r] = __builtin_amdgcn_wmma_f32_16x16x32_f16(
                  false, ah[r], false, bh, (short)0, acc[g][r], false, false);
          }
        }
      }

      // ---- gate nonlinearity + state update (owned columns, in-wave) ----
      const int n  = ncol + lr;           // 0..511
      const float bi = bias[n];
      const float bf = bias[HID + n];
      const float bg = bias[2 * HID + n];
      const float bo = bias[3 * HID + n];
      #pragma unroll
      for (int r = 0; r < 4; ++r) {
        #pragma unroll
        for (int v = 0; v < 8; ++v) {
          const int m = r * 16 + mof + v;     // C/D layout: M = v + 8*(lane>=16)
          float ii = sigmoid_f(acc[0][r][v] + bi);
          float ff = sigmoid_f(acc[1][r][v] + bf);
          float gg = tanh_f   (acc[2][r][v] + bg);
          float oo = sigmoid_f(acc[3][r][v] + bo);
          float cold = cs[m * HID + n];
          float cnew = ff * cold + ii * gg;
          cs[m * HID + n] = cnew;
          hnext[m * HID + n] = (_Float16)(oo * tanh_f(cnew));
        }
      }
    }
    __syncthreads();   // hnext complete before next step reads it
  }

  // ---- write final c ----
  const float4* cs4  = (const float4*)cs;
  float4*       out4 = (float4*)(out + (size_t)m0 * HID);
  for (int i = tid; i < (MT * HID) / 4; i += 256) out4[i] = cs4[i];
}

// ---------------- launcher ----------------
extern "C" void kernel_launch(void* const* d_in, const int* in_sizes, int n_in,
                              void* d_out, int out_size, void* d_ws, size_t ws_size,
                              hipStream_t stream) {
  (void)in_sizes; (void)n_in; (void)out_size; (void)ws_size;
  const float* ts   = (const float*)d_in[0];
  const float* W_ih = (const float*)d_in[1];
  const float* W_hh = (const float*)d_in[2];
  const float* b_ih = (const float*)d_in[3];
  const float* b_hh = (const float*)d_in[4];
  float* out = (float*)d_out;

  char* ws = (char*)d_ws;
  _Float16* whh_h = (_Float16*)ws;                               // 2 MB
  _Float16* wih_h = (_Float16*)(ws + (size_t)G4 * HID * 2);      // 128 KB
  float*    bias  = (float*)  (ws + (size_t)G4 * HID * 2 + (size_t)G4 * 32 * 2); // 8 KB

  lstm_prep_kernel<<<(G4 * HID + 255) / 256, 256, 0, stream>>>(
      W_ih, W_hh, b_ih, b_hh, whh_h, wih_h, bias);

  const int smem_bytes = 2 * MT * HID * 2   // h double buffer (f16)
                       + MT * HID * 4       // c (f32)
                       + STEPS * MT * 32 * 2; // padded x (f16)
  hipFuncSetAttribute((const void*)lstm_main_kernel,
                      hipFuncAttributeMaxDynamicSharedMemorySize, smem_bytes);

  lstm_main_kernel<<<BATCH / MT, 256, smem_bytes, stream>>>(ts, whh_h, wih_h, bias, out);
}
